// IterativeGNNModel_3470333575256
// MI455X (gfx1250) — matrix operand order repaired and checked by reading, not compiled, not run on hardware
//
#include <hip/hip_runtime.h>
#include <math.h>

// ---------------------------------------------------------------------------
// MI455X (gfx1250, wave32) fused GNN forward.
//
// Per-scene transformer conv mapped to 16x16 WMMA tiles:
//   Q,K : [16 x Kp] x [Kp x 512]  -> v_wmma_f32_16x16x32_f16, kept in LDS (f16)
//   V,S : [16 x Kp] x [Kp x dout] -> v_wmma_f32_16x16x32_f16, kept in LDS (f32)
//   Scr : Q x K^T ([16x512]x[512x16]) -> 16 k-steps of wmma -> 16x16 scores
//   masked row-softmax + [16x16]x[16xdout] aggregation, write skip+agg.
// The A panel (16 x Kp activations) is cached in VGPRs per wave and reused
// across all Q/K/V/skip tile jobs; the wave id is readfirstlane'd so job
// selection/branching is scalar, and the tile-store path branches on the
// (scalar) job kind once per tile with the bias hoisted out of the store loop.
// BatchNorm over all 16384 rows runs as separate column-stats kernels.
// ---------------------------------------------------------------------------

typedef __attribute__((ext_vector_type(16))) _Float16 v16h;
typedef __attribute__((ext_vector_type(8)))  _Float16 h8;
typedef __attribute__((ext_vector_type(8)))  float    v8f;

#define NSC    1024          // scenes
#define NPS    16            // nodes per scene
#define NVEH   8
#define NNODES (NSC * NPS)   // 16384
#define KQDIM  512
#define H0D    80
#define H1D    160

// ---- WMMA operand packing (layouts per CDNA5 ISA 7.12.2) --------
// A 16x32 f16: lane m holds row M=m; half0 K={kb..kb+7, kb+16..kb+23} kb=0,
// half1 kb=8.  B 32x16 f16 (column-major storage = W^T row-major): lane n
// holds column N=n; same K order as A (half0 K={0..7,16..23}, half1 K={8..15,24..31}).
static __device__ __forceinline__ v16h packA(const _Float16* row, int kt, int half) {
  const int kb = kt * 32 + half * 8;
  h8 lo = *(const h8*)(row + kb);
  h8 hi = *(const h8*)(row + kb + 16);
  v16h r;
#pragma unroll
  for (int e = 0; e < 8; ++e) { r[e] = lo[e]; r[8 + e] = hi[e]; }
  return r;
}
static __device__ __forceinline__ v16h packB(const _Float16* row, int kt, int half) {
  const int kb = kt * 32 + half * 8;
  h8 lo = *(const h8*)(row + kb);
  h8 hi = *(const h8*)(row + kb + 16);
  v16h r;
#pragma unroll
  for (int e = 0; e < 8; ++e) { r[e] = lo[e]; r[8 + e] = hi[e]; }
  return r;
}

// ---- geometric + structural edge mask, one scene per block -----------------
__global__ __launch_bounds__(256) void k_mask(const float* __restrict__ x0,
                                              float* __restrict__ maskg) {
  const int scene = blockIdx.x;
  const int q = threadIdx.x;          // q = i*16 + j  (i = dst, j = src)
  const int i = q >> 4, j = q & 15;
  float m = 0.0f;
  if (i < NVEH && j != i) {           // structural: dst is vehicle, src != dst
    const float* xd = x0 + (size_t)(scene * NPS + i) * 8;
    const float* xs = x0 + (size_t)(scene * NPS + j) * 8;
    const float dx = xd[0] - xs[0], dy = xd[1] - xs[1];
    const float dist = sqrtf(dx * dx + dy * dy);
    const float veh_e = (xs[7] == 0.0f) ? 1.0f : 0.0f;
    const float margin = fabsf(xd[3]) + veh_e * fabsf(xs[3]);
    const float thr = margin + 2.0f * 1.5f + xs[7] + (veh_e + 1.0f) * 1.5f;
    m = (dist <= thr) ? 1.0f : 0.0f;
  }
  maskg[(size_t)scene * 256 + q] = m;
}

// ---- fp32 [K][N] weight -> f16 transposed+padded [N][Kp] -------------------
__global__ __launch_bounds__(256) void k_w2h(const float* __restrict__ W,
                                             _Float16* __restrict__ Wt,
                                             int K, int Nn, int Kp) {
  const int t = blockIdx.x * 256 + threadIdx.x;
  if (t >= Nn * Kp) return;
  const int nn = t / Kp, k = t % Kp;
  Wt[t] = (k < K) ? (_Float16)W[(size_t)k * Nn + nn] : (_Float16)0.0f;
}

// ---- small dense GEMM Y = X W + b (block0: K=8, block3: N=2) ---------------
__global__ __launch_bounds__(256) void k_gemm_small(const float* __restrict__ X,
                                                    const float* __restrict__ W,
                                                    const float* __restrict__ b,
                                                    float* __restrict__ Y,
                                                    int n, int din, int dout) {
  const int t = blockIdx.x * 256 + threadIdx.x;
  if (t >= n * dout) return;
  const int r = t / dout, c = t % dout;
  float s = b[c];
  for (int k = 0; k < din; ++k) s += X[(size_t)r * din + k] * W[(size_t)k * dout + c];
  Y[t] = s;
}

// ---- per-column BN stats -> (scale, shift) ---------------------------------
__global__ __launch_bounds__(256) void k_stats(const float* __restrict__ Y,
                                               int n, int d,
                                               const float* __restrict__ g,
                                               const float* __restrict__ beta,
                                               float* __restrict__ sc,
                                               float* __restrict__ sh) {
  const int c = blockIdx.x;
  __shared__ float s1[256], s2[256];
  float a = 0.0f, b = 0.0f;
  for (int r = threadIdx.x; r < n; r += 256) {
    const float v = Y[(size_t)r * d + c];
    a += v; b += v * v;
  }
  s1[threadIdx.x] = a; s2[threadIdx.x] = b;
  __syncthreads();
  for (int o = 128; o > 0; o >>= 1) {
    if (threadIdx.x < o) { s1[threadIdx.x] += s1[threadIdx.x + o]; s2[threadIdx.x] += s2[threadIdx.x + o]; }
    __syncthreads();
  }
  if (threadIdx.x == 0) {
    const float mu = s1[0] / (float)n;
    const float var = s2[0] / (float)n - mu * mu;
    const float scale = g[c] * rsqrtf(var + 1e-5f);
    sc[c] = scale; sh[c] = beta[c] - mu * scale;
  }
}

__global__ __launch_bounds__(256) void k_bn_relu(const float* __restrict__ Y,
                                                 const float* __restrict__ sc,
                                                 const float* __restrict__ sh,
                                                 float* __restrict__ X, int n, int d) {
  const int t = blockIdx.x * 256 + threadIdx.x;
  if (t >= n * d) return;
  const int c = t % d;
  X[t] = fmaxf(sc[c] * Y[t] + sh[c], 0.0f);
}

__global__ __launch_bounds__(256) void k_res(const float* __restrict__ Y,
                                             const float* __restrict__ sc,
                                             const float* __restrict__ sh,
                                             float* __restrict__ X, int n, int d) {
  const int t = blockIdx.x * 256 + threadIdx.x;
  if (t >= n * d) return;
  const int c = t % d;
  X[t] = fmaxf(sc[c] * Y[t] + sh[c] + X[t], 0.0f);
}

// ---- fused transformer conv: one scene per 256-thread (8-wave) block -------
// KP (padded K, multiple of 32) and DOUT are compile-time so the per-lane A
// fragments stay in VGPRs across all tile jobs.
template <int KP, int DOUT>
__global__ __launch_bounds__(256) void k_conv(const float* __restrict__ Xin,
                                              int din,
                                              const _Float16* __restrict__ WqT,
                                              const _Float16* __restrict__ WkT,
                                              const _Float16* __restrict__ WvT,
                                              const _Float16* __restrict__ WsT,
                                              const float* __restrict__ bV,
                                              const float* __restrict__ bS,
                                              const float* __restrict__ maskg,
                                              float* __restrict__ Yout) {
  constexpr int KS  = KP / 32;     // k-steps for projections
  constexpr int NTV = DOUT / 16;   // N-tiles for V / skip
  constexpr int JOBS = 64 + 2 * NTV;

  __shared__ __align__(16) _Float16 sA[NPS * KP];     // f16 activations (padded)
  __shared__ __align__(16) _Float16 sQ[NPS * KQDIM];
  __shared__ __align__(16) _Float16 sK[NPS * KQDIM];
  __shared__ __align__(16) float    sV[NPS * DOUT];
  __shared__ __align__(16) float    sSk[NPS * DOUT];
  __shared__ float sScore[NPS * NPS];
  __shared__ float sW[NPS * NPS];
  __shared__ float sDen[NPS];

  const int scene = blockIdx.x;
  const int tid = threadIdx.x;
  // wave id is uniform across the wave32 -- assert it so job selection and
  // branching lower to scalar (s_cmp/s_cbranch), not exec-mask chains.
  const int wave = __builtin_amdgcn_readfirstlane(tid >> 5);
  const int lane = tid & 31;
  const int half = lane >> 4, mr = lane & 15;

  // stage activations -> LDS f16, zero-padded to KP
  for (int t = tid; t < NPS * KP; t += 256) {
    const int m = t / KP, c = t % KP;
    sA[t] = (_Float16)((c < din) ? Xin[(size_t)(scene * NPS + m) * din + c] : 0.0f);
  }
  __syncthreads();

  // hoist the A panel into registers once per wave (reused by every tile job)
  v16h afrag[KS];
  {
    const _Float16* Arow = sA + mr * KP;
#pragma unroll
    for (int kt = 0; kt < KS; ++kt) afrag[kt] = packA(Arow, kt, half);
  }

  // tile jobs: 32 Q-tiles, 32 K-tiles, NTV V-tiles, NTV skip-tiles
  for (int j = wave; j < JOBS; j += 8) {
    int kind, nt;
    const _Float16* Bt;
    if (j < 32)            { kind = 0; nt = j;            Bt = WqT; }
    else if (j < 64)       { kind = 1; nt = j - 32;       Bt = WkT; }
    else if (j < 64 + NTV) { kind = 2; nt = j - 64;       Bt = WvT; }
    else                   { kind = 3; nt = j - 64 - NTV; Bt = WsT; }
    v8f acc = {0.f, 0.f, 0.f, 0.f, 0.f, 0.f, 0.f, 0.f};
    const _Float16* Brow = Bt + (size_t)(nt * 16 + mr) * KP;
#pragma unroll
    for (int kt = 0; kt < KS; ++kt) {
      const v16h bv = packB(Brow, kt, half);
      acc = __builtin_amdgcn_wmma_f32_16x16x32_f16(false, afrag[kt], false, bv,
                                                   (short)0, acc, false, false);
    }
    // store tile: branch on scalar kind ONCE, bias hoisted out of the r-loop
    const int n = nt * 16 + mr;
    const int mb = half * 8;
    if (kind == 0) {
#pragma unroll
      for (int r = 0; r < 8; ++r) sQ[(r + mb) * KQDIM + n] = (_Float16)acc[r];
    } else if (kind == 1) {
#pragma unroll
      for (int r = 0; r < 8; ++r) sK[(r + mb) * KQDIM + n] = (_Float16)acc[r];
    } else if (kind == 2) {
      const float bias = bV[n];
#pragma unroll
      for (int r = 0; r < 8; ++r) sV[(r + mb) * DOUT + n] = acc[r] + bias;
    } else {
      const float bias = bS[n];
#pragma unroll
      for (int r = 0; r < 8; ++r) sSk[(r + mb) * DOUT + n] = acc[r] + bias;
    }
  }
  __syncthreads();

  // scores S = Q K^T (16x16), wave 0: 16 k-steps over KQ=512.
  // K stored row-major [16][512] == column-major B operand (lane = src column).
  if (wave == 0) {
    v8f acc = {0.f, 0.f, 0.f, 0.f, 0.f, 0.f, 0.f, 0.f};
    const _Float16* Arow = sQ + mr * KQDIM;
    const _Float16* Brow = sK + mr * KQDIM;
#pragma unroll
    for (int kt = 0; kt < KQDIM / 32; ++kt) {
      const v16h av = packA(Arow, kt, half);
      const v16h bv = packB(Brow, kt, half);
      acc = __builtin_amdgcn_wmma_f32_16x16x32_f16(false, av, false, bv,
                                                   (short)0, acc, false, false);
    }
    const float rs = 0.04419417382415922f;  // 1/sqrt(512)
#pragma unroll
    for (int r = 0; r < 8; ++r)
      sScore[(r + half * 8) * NPS + mr] = acc[r] * rs;
  }
  __syncthreads();

  // masked row-softmax over incoming edges (row i = dst node)
  if (tid < NPS) {
    const int i = tid;
    const float* mrow = maskg + (size_t)scene * 256 + i * NPS;
    float smax = -3.0e38f;
#pragma unroll
    for (int j = 0; j < NPS; ++j)
      if (mrow[j] > 0.0f) smax = fmaxf(smax, sScore[i * NPS + j]);
    float den = 0.0f;
#pragma unroll
    for (int j = 0; j < NPS; ++j) {
      const float w = (mrow[j] > 0.0f) ? __expf(sScore[i * NPS + j] - smax) : 0.0f;
      sW[i * NPS + j] = w;
      den += w;
    }
    sDen[i] = den;
  }
  __syncthreads();

  // out = skip + (softmax @ V) / (den + eps)
  for (int t = tid; t < NPS * DOUT; t += 256) {
    const int i = t / DOUT, c = t % DOUT;
    float a = 0.0f;
#pragma unroll
    for (int j = 0; j < NPS; ++j) a += sW[i * NPS + j] * sV[j * DOUT + c];
    Yout[(size_t)(scene * NPS + i) * DOUT + c] = sSk[t] + a / (sDen[i] + 1e-16f);
  }
}

// ---- final: BN + tanh + bound, gather vehicle rows -------------------------
__global__ __launch_bounds__(256) void k_final(const float* __restrict__ Y3,
                                               const float* __restrict__ sc,
                                               const float* __restrict__ sh,
                                               float* __restrict__ out) {
  const int t = blockIdx.x * 256 + threadIdx.x;
  if (t >= NSC * NVEH * 2) return;
  const int r = t / 2, o = t % 2;
  const int node = (r / NVEH) * NPS + (r % NVEH);
  const float bound = (o == 0) ? 1.0f : 0.8f;
  out[t] = tanhf(sc[o] * Y3[(size_t)node * 2 + o] + sh[o]) * bound;
}

// ---------------------------------------------------------------------------
extern "C" void kernel_launch(void* const* d_in, const int* in_sizes, int n_in,
                              void* d_out, int out_size, void* d_ws, size_t ws_size,
                              hipStream_t stream) {
  (void)in_sizes; (void)n_in; (void)out_size; (void)ws_size;
  const float* x0 = (const float*)d_in[0];
  // d_in[1] batches / d_in[2] edges / d_in[3] veh_indices: topology is fixed
  // (dst = vehicles 0..7, src = all 15 other nodes per scene) -> implicit.

  // Params flattened in JAX pytree order (dict keys sorted):
  // block0: W,b,beta,g ; blockN: b1,b2,conv1{Wk,Wq,Ws,Wv,bs,bv},conv2{...},g1,g2
  int pi = 4;
  auto P = [&]() -> const float* { return (const float*)d_in[pi++]; };
  const float *W0 = P(), *b0 = P(), *beta0 = P(), *g0 = P();
  struct ConvP { const float *Wk, *Wq, *Ws, *Wv, *bs, *bv; };
  struct ResP  { const float *b1, *b2, *g1, *g2; ConvP c1, c2; };
  auto parseConv = [&](ConvP& c) { c.Wk = P(); c.Wq = P(); c.Ws = P(); c.Wv = P(); c.bs = P(); c.bv = P(); };
  auto parseRes  = [&](ResP& r)  { r.b1 = P(); r.b2 = P(); parseConv(r.c1); parseConv(r.c2); r.g1 = P(); r.g2 = P(); };
  ResP r1, r2; parseRes(r1); parseRes(r2);
  const float *W3 = P(), *b3 = P(), *beta3 = P(), *g3 = P();

  // workspace bump allocator (256B aligned)
  size_t off = 0;
  auto alloc = [&](size_t bytes) -> void* {
    void* p = (void*)((char*)d_ws + off);
    off += (bytes + 255) & ~(size_t)255;
    return p;
  };
  float* mask = (float*)alloc((size_t)NSC * 256 * 4);
  float* X    = (float*)alloc((size_t)NNODES * H0D * 4);
  float* Y1   = (float*)alloc((size_t)NNODES * H1D * 4);
  float* Y2   = (float*)alloc((size_t)NNODES * H0D * 4);
  float* T1   = (float*)alloc((size_t)NNODES * H1D * 4);
  float* Y3   = (float*)alloc((size_t)NNODES * 2 * 4);
  float* sc   = (float*)alloc(1024);
  float* sh   = (float*)alloc(1024);
  struct ConvH { _Float16 *Wq, *Wk, *Wv, *Ws; };
  auto allocConvH = [&](int Kp, int dout, ConvH& h) {
    h.Wq = (_Float16*)alloc((size_t)KQDIM * Kp * 2);
    h.Wk = (_Float16*)alloc((size_t)KQDIM * Kp * 2);
    h.Wv = (_Float16*)alloc((size_t)dout * Kp * 2);
    h.Ws = (_Float16*)alloc((size_t)dout * Kp * 2);
  };
  ConvH h11, h12, h21, h22;
  allocConvH(96, H1D, h11);  allocConvH(160, H0D, h12);
  allocConvH(96, H1D, h21);  allocConvH(160, H0D, h22);

  auto w2h = [&](const float* W, _Float16* Wt, int K, int Nn, int Kp) {
    const int tot = Nn * Kp;
    k_w2h<<<(tot + 255) / 256, 256, 0, stream>>>(W, Wt, K, Nn, Kp);
  };
  auto cvtConv = [&](const ConvP& c, const ConvH& h, int din, int Kp, int dout) {
    w2h(c.Wq, h.Wq, din, KQDIM, Kp);
    w2h(c.Wk, h.Wk, din, KQDIM, Kp);
    w2h(c.Wv, h.Wv, din, dout, Kp);
    w2h(c.Ws, h.Ws, din, dout, Kp);
  };

  // edge mask + weight conversion
  k_mask<<<NSC, 256, 0, stream>>>(x0, mask);
  cvtConv(r1.c1, h11, H0D, 96, H1D);  cvtConv(r1.c2, h12, H1D, 160, H0D);
  cvtConv(r2.c1, h21, H0D, 96, H1D);  cvtConv(r2.c2, h22, H1D, 160, H0D);

  // block0: [16384x8]x[8x80] + BN + relu
  k_gemm_small<<<(NNODES * H0D + 255) / 256, 256, 0, stream>>>(x0, W0, b0, Y2, NNODES, 8, H0D);
  k_stats<<<H0D, 256, 0, stream>>>(Y2, NNODES, H0D, g0, beta0, sc, sh);
  k_bn_relu<<<(NNODES * H0D + 255) / 256, 256, 0, stream>>>(Y2, sc, sh, X, NNODES, H0D);

  // residual blocks
  auto resblock = [&](const ResP& r, const ConvH& hA, const ConvH& hB) {
    k_conv<96, H1D><<<NSC, 256, 0, stream>>>(X, H0D, hA.Wq, hA.Wk, hA.Wv, hA.Ws,
                                             r.c1.bv, r.c1.bs, mask, Y1);
    k_stats<<<H1D, 256, 0, stream>>>(Y1, NNODES, H1D, r.g1, r.b1, sc, sh);
    k_bn_relu<<<(NNODES * H1D + 255) / 256, 256, 0, stream>>>(Y1, sc, sh, T1, NNODES, H1D);
    k_conv<160, H0D><<<NSC, 256, 0, stream>>>(T1, H1D, hB.Wq, hB.Wk, hB.Wv, hB.Ws,
                                              r.c2.bv, r.c2.bs, mask, Y2);
    k_stats<<<H0D, 256, 0, stream>>>(Y2, NNODES, H0D, r.g2, r.b2, sc, sh);
    k_res<<<(NNODES * H0D + 255) / 256, 256, 0, stream>>>(Y2, sc, sh, X, NNODES, H0D);
  };
  resblock(r1, h11, h12);
  resblock(r2, h21, h22);

  // block3: [16384x80]x[80x2] + BN + tanh*bound, gather vehicles
  k_gemm_small<<<(NNODES * 2 + 255) / 256, 256, 0, stream>>>(X, W3, b3, Y3, NNODES, H0D, 2);
  k_stats<<<2, 256, 0, stream>>>(Y3, NNODES, 2, g3, beta3, sc, sh);
  k_final<<<(NSC * NVEH * 2 + 255) / 256, 256, 0, stream>>>(Y3, sc, sh, (float*)d_out);
}
